// HedgehogAttention_15522011808308
// MI455X (gfx1250) — compile-verified, matrix operand order checked by reading, not run on hardware
//
#include <hip/hip_runtime.h>
#include <hip/hip_bf16.h>
#include <math.h>

typedef _Float16 half_t;
typedef __attribute__((ext_vector_type(16))) _Float16 v16h;
typedef __attribute__((ext_vector_type(8)))  _Float16 v8h;
typedef __attribute__((ext_vector_type(8)))  float    v8f;

#define BB     2
#define SS     1024
#define DMODEL 1024
#define NH     16
#define HD     64

// ---------------------------------------------------------------------------
// Fragment helpers for V_WMMA_F32_16X16X32_F16 (wave32).
// A operand: lane L holds row (L%16); K-halves are {g*8 .. g*8+7} and
// {16+g*8 .. 16+g*8+7} with g = L/16  -> two contiguous 16B loads per lane.
// B operand: identical layout over rows of B^T.
// ---------------------------------------------------------------------------
__device__ __forceinline__ v16h load_frag_h(const half_t* __restrict__ base,
                                            int row, int k0, int ld) {
  const int lane = threadIdx.x & 31;
  const half_t* p = base + (size_t)(row + (lane & 15)) * ld
                         + (k0 + ((lane >> 4) << 3));
  v8h lo = *(const v8h*)(p);
  v8h hi = *(const v8h*)(p + 16);
  v16h out;
#pragma unroll
  for (int i = 0; i < 8; ++i) { out[i] = lo[i]; out[8 + i] = hi[i]; }
  return out;
}

__device__ __forceinline__ v16h load_frag_f(const float* __restrict__ base,
                                            int row, int k0, int ld) {
  const int lane = threadIdx.x & 31;
  const float* p = base + (size_t)(row + (lane & 15)) * ld
                        + (k0 + ((lane >> 4) << 3));
  v16h out;
#pragma unroll
  for (int i = 0; i < 8; ++i) {
    out[i]     = (half_t)p[i];
    out[8 + i] = (half_t)p[16 + i];
  }
  return out;
}

#define WMMA(a, b, c) \
  __builtin_amdgcn_wmma_f32_16x16x32_f16(false, (a), false, (b), (short)0, (c), false, false)

// ---------------------------------------------------------------------------
// f32 -> f16 staging
// ---------------------------------------------------------------------------
__global__ void cvt_f32_f16_kernel(const float* __restrict__ in,
                                   half_t* __restrict__ out, int n) {
  int i = blockIdx.x * blockDim.x + threadIdx.x;
  if (i < n) out[i] = (half_t)in[i];
}

// ---------------------------------------------------------------------------
// QKV projection, 2x2 register-blocked (32x32 per wave):
// C[m,n] = sum_k h16[m,k]*W[n,k] + bias[n]
// z=0 -> q16 (B,H,S,D), z=1 -> k16 (B,H,S,D), z=2 -> vT16 (B,H,D,S)
// ---------------------------------------------------------------------------
__device__ __forceinline__ void store_qkv_tile(int z, int tm, int tn, v8f acc,
                                               half_t* __restrict__ q16,
                                               half_t* __restrict__ k16,
                                               half_t* __restrict__ vT16) {
  const int lane = threadIdx.x & 31;
  const int n  = tn + (lane & 15);
  const int hh = n >> 6, dd = n & 63;
  const int m0 = tm + ((lane >> 4) << 3);
#pragma unroll
  for (int r = 0; r < 8; ++r) {
    const int m  = m0 + r;
    const int bb = m >> 10, ss = m & 1023;
    const half_t val = (half_t)acc[r];
    if (z == 0)
      q16[((size_t)(bb * NH + hh) * SS + ss) * HD + dd] = val;
    else if (z == 1)
      k16[((size_t)(bb * NH + hh) * SS + ss) * HD + dd] = val;
    else
      vT16[((size_t)(bb * NH + hh) * HD + dd) * SS + ss] = val;   // transposed
  }
}

__global__ void proj_qkv_kernel(const half_t* __restrict__ h16,
                                const half_t* __restrict__ wq,
                                const half_t* __restrict__ wk,
                                const half_t* __restrict__ wv,
                                const float* __restrict__ bq,
                                const float* __restrict__ bk,
                                const float* __restrict__ bv,
                                half_t* __restrict__ q16,
                                half_t* __restrict__ k16,
                                half_t* __restrict__ vT16) {
  const int wave = threadIdx.x >> 5;
  const int lane = threadIdx.x & 31;
  const int tm = (blockIdx.x * 4 + wave) * 32;
  const int tn = blockIdx.y * 32;
  const int z  = blockIdx.z;
  const half_t* W    = (z == 0) ? wq : (z == 1) ? wk : wv;
  const float*  bias = (z == 0) ? bq : (z == 1) ? bk : bv;

  const float bc0 = bias[tn + (lane & 15)];
  const float bc1 = bias[tn + 16 + (lane & 15)];
  v8f acc00, acc01, acc10, acc11;
#pragma unroll
  for (int i = 0; i < 8; ++i) {
    acc00[i] = bc0; acc10[i] = bc0;
    acc01[i] = bc1; acc11[i] = bc1;
  }

  for (int kk = 0; kk < DMODEL; kk += 32) {
    v16h a0 = load_frag_h(h16, tm,      kk, DMODEL);
    v16h a1 = load_frag_h(h16, tm + 16, kk, DMODEL);
    v16h b0 = load_frag_h(W,   tn,      kk, DMODEL);
    v16h b1 = load_frag_h(W,   tn + 16, kk, DMODEL);
    acc00 = WMMA(a0, b0, acc00);
    acc01 = WMMA(a0, b1, acc01);
    acc10 = WMMA(a1, b0, acc10);
    acc11 = WMMA(a1, b1, acc11);
  }

  store_qkv_tile(z, tm,      tn,      acc00, q16, k16, vT16);
  store_qkv_tile(z, tm,      tn + 16, acc01, q16, k16, vT16);
  store_qkv_tile(z, tm + 16, tn,      acc10, q16, k16, vT16);
  store_qkv_tile(z, tm + 16, tn + 16, acc11, q16, k16, vT16);
}

// ---------------------------------------------------------------------------
// Per-(b,h) "A @ B^T" scores, 2x2 register-blocked:
// out[z][m][n] = scale * sum_k A[z][m][k]*B[z][n][k]
// kdim = 64 (true attention, scale 1/8) or 128 (hedgehog, scale 1)
// ---------------------------------------------------------------------------
__global__ void scores_kernel(const half_t* __restrict__ A,
                              const half_t* __restrict__ B,
                              float* __restrict__ out,
                              int kdim, float scale) {
  const int wave = threadIdx.x >> 5;
  const int lane = threadIdx.x & 31;
  const int tm = (blockIdx.x * 4 + wave) * 32;
  const int tn = blockIdx.y * 32;
  const int z  = blockIdx.z;
  const half_t* Az = A + (size_t)z * SS * kdim;
  const half_t* Bz = B + (size_t)z * SS * kdim;
  float*        Oz = out + (size_t)z * SS * SS;

  v8f acc00 = {}, acc01 = {}, acc10 = {}, acc11 = {};
  for (int kk = 0; kk < kdim; kk += 32) {
    v16h a0 = load_frag_h(Az, tm,      kk, kdim);
    v16h a1 = load_frag_h(Az, tm + 16, kk, kdim);
    v16h b0 = load_frag_h(Bz, tn,      kk, kdim);
    v16h b1 = load_frag_h(Bz, tn + 16, kk, kdim);
    acc00 = WMMA(a0, b0, acc00);
    acc01 = WMMA(a0, b1, acc01);
    acc10 = WMMA(a1, b0, acc10);
    acc11 = WMMA(a1, b1, acc11);
  }

  auto st = [&](v8f acc, int rm, int rn) {
    const int n  = rn + (lane & 15);
    const int m0 = rm + ((lane >> 4) << 3);
#pragma unroll
    for (int r = 0; r < 8; ++r)
      Oz[(size_t)(m0 + r) * SS + n] = acc[r] * scale;
  };
  st(acc00, tm,      tn);
  st(acc01, tm,      tn + 16);
  st(acc10, tm + 16, tn);
  st(acc11, tm + 16, tn + 16);
}

// ---------------------------------------------------------------------------
// Row softmax, in place
// ---------------------------------------------------------------------------
__global__ void softmax_rows_kernel(float* __restrict__ data, int n) {
  __shared__ float red[256];
  float* row = data + (size_t)blockIdx.x * n;
  const int t = threadIdx.x;

  float m = -3.4e38f;
  for (int i = t; i < n; i += 256) m = fmaxf(m, row[i]);
  red[t] = m; __syncthreads();
  for (int s = 128; s > 0; s >>= 1) {
    if (t < s) red[t] = fmaxf(red[t], red[t + s]);
    __syncthreads();
  }
  const float rmax = red[0]; __syncthreads();

  float sum = 0.f;
  for (int i = t; i < n; i += 256) {
    float e = __expf(row[i] - rmax);
    row[i] = e;
    sum += e;
  }
  red[t] = sum; __syncthreads();
  for (int s = 128; s > 0; s >>= 1) {
    if (t < s) red[t] += red[t + s];
    __syncthreads();
  }
  const float inv = 1.0f / red[0];
  for (int i = t; i < n; i += 256) row[i] *= inv;
}

// Row sum-normalize, in place
__global__ void rowdiv_kernel(float* __restrict__ data, int n) {
  __shared__ float red[256];
  float* row = data + (size_t)blockIdx.x * n;
  const int t = threadIdx.x;
  float sum = 0.f;
  for (int i = t; i < n; i += 256) sum += row[i];
  red[t] = sum; __syncthreads();
  for (int s = 128; s > 0; s >>= 1) {
    if (t < s) red[t] += red[t + s];
    __syncthreads();
  }
  const float inv = 1.0f / red[0];
  for (int i = t; i < n; i += 256) row[i] *= inv;
}

// ---------------------------------------------------------------------------
// ctx[z][s][d] = sum_n attn[z][s][n] * V[z][n][d], via V^T operand (2x2 tiles)
// ---------------------------------------------------------------------------
__global__ void ctx_kernel(const float* __restrict__ attn,
                           const half_t* __restrict__ vT,
                           half_t* __restrict__ ctx16) {
  const int wave = threadIdx.x >> 5;
  const int lane = threadIdx.x & 31;
  const int tm = (blockIdx.x * 4 + wave) * 32;   // s tile
  const int tn = blockIdx.y * 32;                // d tile
  const int z  = blockIdx.z;
  const float*  Az = attn + (size_t)z * SS * SS;
  const half_t* Bz = vT + (size_t)z * HD * SS;

  v8f acc00 = {}, acc01 = {}, acc10 = {}, acc11 = {};
  for (int kk = 0; kk < SS; kk += 32) {
    v16h a0 = load_frag_f(Az, tm,      kk, SS);
    v16h a1 = load_frag_f(Az, tm + 16, kk, SS);
    v16h b0 = load_frag_h(Bz, tn,      kk, SS);
    v16h b1 = load_frag_h(Bz, tn + 16, kk, SS);
    acc00 = WMMA(a0, b0, acc00);
    acc01 = WMMA(a0, b1, acc01);
    acc10 = WMMA(a1, b0, acc10);
    acc11 = WMMA(a1, b1, acc11);
  }

  const int bb = z >> 4, hh = z & 15;
  auto st = [&](v8f acc, int rm, int rn) {
    const int d  = rn + (lane & 15);
    const int m0 = rm + ((lane >> 4) << 3);
#pragma unroll
    for (int r = 0; r < 8; ++r) {
      const int s = m0 + r;
      ctx16[(size_t)(bb * SS + s) * DMODEL + hh * HD + d] = (half_t)acc[r];
    }
  };
  st(acc00, tm,      tn);
  st(acc01, tm,      tn + 16);
  st(acc10, tm + 16, tn);
  st(acc11, tm + 16, tn + 16);
}

// ---------------------------------------------------------------------------
// Output projection (2x2): out[m,n] = sum_k ctx16[m,k]*Wo[n,k] + bo[n]
// ---------------------------------------------------------------------------
__global__ void outproj_kernel(const half_t* __restrict__ ctx16,
                               const half_t* __restrict__ wo,
                               const float* __restrict__ bo,
                               float* __restrict__ out) {
  const int wave = threadIdx.x >> 5;
  const int lane = threadIdx.x & 31;
  const int tm = (blockIdx.x * 4 + wave) * 32;
  const int tn = blockIdx.y * 32;

  const float bc0 = bo[tn + (lane & 15)];
  const float bc1 = bo[tn + 16 + (lane & 15)];
  v8f acc00, acc01, acc10, acc11;
#pragma unroll
  for (int i = 0; i < 8; ++i) {
    acc00[i] = bc0; acc10[i] = bc0;
    acc01[i] = bc1; acc11[i] = bc1;
  }

  for (int kk = 0; kk < DMODEL; kk += 32) {
    v16h a0 = load_frag_h(ctx16, tm,      kk, DMODEL);
    v16h a1 = load_frag_h(ctx16, tm + 16, kk, DMODEL);
    v16h b0 = load_frag_h(wo,    tn,      kk, DMODEL);
    v16h b1 = load_frag_h(wo,    tn + 16, kk, DMODEL);
    acc00 = WMMA(a0, b0, acc00);
    acc01 = WMMA(a0, b1, acc01);
    acc10 = WMMA(a1, b0, acc10);
    acc11 = WMMA(a1, b1, acc11);
  }

  auto st = [&](v8f acc, int rm, int rn) {
    const int n  = rn + (lane & 15);
    const int m0 = rm + ((lane >> 4) << 3);
#pragma unroll
    for (int r = 0; r < 8; ++r)
      out[(size_t)(m0 + r) * DMODEL + n] = acc[r];
  };
  st(acc00, tm,      tn);
  st(acc01, tm,      tn + 16);
  st(acc10, tm + 16, tn);
  st(acc11, tm + 16, tn + 16);
}

// ---------------------------------------------------------------------------
// Hedgehog feature maps (2x2): y = x @ Wf^T + bf (64x64 per head),
// f[..., e] = exp(y_e), f[..., 64+e] = exp(-y_e).  M = B*H*S = 32768 rows.
// ---------------------------------------------------------------------------
__global__ void feature_kernel(const half_t* __restrict__ q16,
                               const half_t* __restrict__ k16,
                               const half_t* __restrict__ wfq,
                               const half_t* __restrict__ wfk,
                               const float* __restrict__ bfq,
                               const float* __restrict__ bfk,
                               half_t* __restrict__ fq16,
                               half_t* __restrict__ fk16) {
  const int wave = threadIdx.x >> 5;
  const int lane = threadIdx.x & 31;
  const int tm = (blockIdx.x * 4 + wave) * 32;   // flat (b,h,s) row tile
  const int tn = blockIdx.y * 32;                // e tile
  const int z  = blockIdx.z;
  const half_t* A    = (z == 0) ? q16 : k16;
  const half_t* W    = (z == 0) ? wfq : wfk;
  const float*  bias = (z == 0) ? bfq : bfk;
  half_t*       out  = (z == 0) ? fq16 : fk16;

  const float bc0 = bias[tn + (lane & 15)];
  const float bc1 = bias[tn + 16 + (lane & 15)];
  v8f acc00, acc01, acc10, acc11;
#pragma unroll
  for (int i = 0; i < 8; ++i) {
    acc00[i] = bc0; acc10[i] = bc0;
    acc01[i] = bc1; acc11[i] = bc1;
  }

  for (int kk = 0; kk < HD; kk += 32) {
    v16h a0 = load_frag_h(A, tm,      kk, HD);
    v16h a1 = load_frag_h(A, tm + 16, kk, HD);
    v16h b0 = load_frag_h(W, tn,      kk, HD);
    v16h b1 = load_frag_h(W, tn + 16, kk, HD);
    acc00 = WMMA(a0, b0, acc00);
    acc01 = WMMA(a0, b1, acc01);
    acc10 = WMMA(a1, b0, acc10);
    acc11 = WMMA(a1, b1, acc11);
  }

  auto st = [&](v8f acc, int rm, int rn) {
    const int e  = rn + (lane & 15);
    const int m0 = rm + ((lane >> 4) << 3);
#pragma unroll
    for (int r = 0; r < 8; ++r) {
      const float y = acc[r];
      half_t* row = out + (size_t)(m0 + r) * (2 * HD);
      row[e]      = (half_t)__expf(y);
      row[HD + e] = (half_t)__expf(-y);
    }
  };
  st(acc00, tm,      tn);
  st(acc01, tm,      tn + 16);
  st(acc10, tm + 16, tn);
  st(acc11, tm + 16, tn + 16);
}

// ---------------------------------------------------------------------------
// Host launch
// ---------------------------------------------------------------------------
extern "C" void kernel_launch(void* const* d_in, const int* in_sizes, int n_in,
                              void* d_out, int out_size, void* d_ws, size_t ws_size,
                              hipStream_t stream) {
  (void)in_sizes; (void)n_in; (void)out_size; (void)ws_size;

  const float* hidden = (const float*)d_in[0];
  const float* Wq  = (const float*)d_in[1];
  const float* bq  = (const float*)d_in[2];
  const float* Wk  = (const float*)d_in[3];
  const float* bk  = (const float*)d_in[4];
  const float* Wv  = (const float*)d_in[5];
  const float* bv  = (const float*)d_in[6];
  const float* Wo  = (const float*)d_in[7];
  const float* bo  = (const float*)d_in[8];
  const float* Wfq = (const float*)d_in[9];
  const float* bfq = (const float*)d_in[10];
  const float* Wfk = (const float*)d_in[11];
  const float* bfk = (const float*)d_in[12];

  float* outp  = (float*)d_out;                           // (B,S,DM)
  float* pred  = outp + (size_t)BB * SS * DMODEL;          // (B,H,S,S)
  float* truea = pred + (size_t)BB * NH * SS * SS;         // (B,H,S,S)

  size_t off = 0;
  auto carve = [&](size_t bytes) {
    void* p = (char*)d_ws + off;
    off += (bytes + 255) & ~(size_t)255;
    return p;
  };
  const size_t NHID = (size_t)BB * SS * DMODEL;            // 2,097,152
  const size_t NW   = (size_t)DMODEL * DMODEL;             // 1,048,576
  half_t* h16   = (half_t*)carve(NHID * 2);
  half_t* wq16  = (half_t*)carve(NW * 2);
  half_t* wk16  = (half_t*)carve(NW * 2);
  half_t* wv16  = (half_t*)carve(NW * 2);
  half_t* wo16  = (half_t*)carve(NW * 2);
  half_t* wfq16 = (half_t*)carve((size_t)HD * HD * 2);
  half_t* wfk16 = (half_t*)carve((size_t)HD * HD * 2);
  half_t* q16   = (half_t*)carve(NHID * 2);                // (B,H,S,D)
  half_t* k16   = (half_t*)carve(NHID * 2);                // (B,H,S,D)
  half_t* vT16  = (half_t*)carve(NHID * 2);                // (B,H,D,S)
  half_t* ctx16 = (half_t*)carve(NHID * 2);                // (B,S,DM)
  half_t* fq16  = (half_t*)carve(NHID * 2 * 2);            // (B,H,S,2D)
  half_t* fk16  = (half_t*)carve(NHID * 2 * 2);            // (B,H,S,2D)

  // 1) f32 -> f16 staging
  cvt_f32_f16_kernel<<<(int)((NHID + 255) / 256), 256, 0, stream>>>(hidden, h16, (int)NHID);
  cvt_f32_f16_kernel<<<(int)((NW + 255) / 256), 256, 0, stream>>>(Wq, wq16, (int)NW);
  cvt_f32_f16_kernel<<<(int)((NW + 255) / 256), 256, 0, stream>>>(Wk, wk16, (int)NW);
  cvt_f32_f16_kernel<<<(int)((NW + 255) / 256), 256, 0, stream>>>(Wv, wv16, (int)NW);
  cvt_f32_f16_kernel<<<(int)((NW + 255) / 256), 256, 0, stream>>>(Wo, wo16, (int)NW);
  cvt_f32_f16_kernel<<<16, 256, 0, stream>>>(Wfq, wfq16, HD * HD);
  cvt_f32_f16_kernel<<<16, 256, 0, stream>>>(Wfk, wfk16, HD * HD);

  // 2) QKV projections (WMMA 2x2)
  proj_qkv_kernel<<<dim3(16, 32, 3), 128, 0, stream>>>(
      h16, wq16, wk16, wv16, bq, bk, bv, q16, k16, vT16);

  // 3) true scores = Q.K^T / sqrt(D)
  scores_kernel<<<dim3(8, 32, BB * NH), 128, 0, stream>>>(
      q16, k16, truea, HD, 0.125f);

  // 4) softmax rows (in place)
  softmax_rows_kernel<<<BB * NH * SS, 256, 0, stream>>>(truea, SS);

  // 5) ctx = attn @ V
  ctx_kernel<<<dim3(8, 2, BB * NH), 128, 0, stream>>>(truea, vT16, ctx16);

  // 6) outputs = ctx @ Wo^T + bo
  outproj_kernel<<<dim3(16, 32), 128, 0, stream>>>(ctx16, wo16, bo, outp);

  // 7) hedgehog feature maps
  feature_kernel<<<dim3(256, 2, 2), 128, 0, stream>>>(
      q16, k16, wfq16, wfk16, bfq, bfk, fq16, fk16);

  // 8) hedgehog scores = fq @ fk^T (K = 128)
  scores_kernel<<<dim3(8, 32, BB * NH), 128, 0, stream>>>(
      fq16, fk16, pred, 2 * HD, 1.0f);

  // 9) row-sum normalize (in place)
  rowdiv_kernel<<<BB * NH * SS, 256, 0, stream>>>(pred, SS);
}
